// Classifier_71829033059182
// MI455X (gfx1250) — matrix operand order repaired and checked by reading, not compiled, not run on hardware
//
#include <hip/hip_runtime.h>
#include <hip/hip_bf16.h>
#include <math.h>
#include <stdint.h>

#define DEV __device__ __forceinline__

typedef __attribute__((ext_vector_type(16))) __bf16 v16bf;
typedef __attribute__((ext_vector_type(8)))  float  v8f;

static constexpr int  Bn = 4096, Kn = 64, Dn = 128, Hn = 2048;
static constexpr int  INn = Dn + Kn * (Dn + 5);   // 8640
static constexpr float EPSf = 1e-5f;

// ---------------- helpers ----------------

DEV unsigned short f2bf(float f) {
  unsigned u = __float_as_uint(f);
  unsigned r = (u + 0x7FFFu + ((u >> 16) & 1u)) >> 16;
  return (unsigned short)r;
}

DEV float wave_reduce_add(float v) {
  v += __shfl_xor(v, 16, 32);
  v += __shfl_xor(v,  8, 32);
  v += __shfl_xor(v,  4, 32);
  v += __shfl_xor(v,  2, 32);
  v += __shfl_xor(v,  1, 32);
  return v;
}

// gfx1250 async global->LDS copy, 16B per lane, tracked by ASYNCcnt.
// lds_off: byte offset in LDS; gvoff: per-lane byte offset; base: uniform SGPR base.
DEV void async_copy_b128(unsigned lds_off, unsigned gvoff, const void* base) {
  asm volatile("global_load_async_to_lds_b128 %0, %1, %2 offset:0"
               :: "v"(lds_off), "v"(gvoff), "s"(base)
               : "memory");
}

template<int N>
DEV void wait_asynccnt() {
  asm volatile("s_wait_asynccnt %0" :: "n"(N) : "memory");
}

// ---------------- weight prep ----------------

// f32 [Krows, N] -> bf16 transposed [N, Krows]
__global__ __launch_bounds__(256) void cvt_transpose_kernel(
    const float* __restrict__ W, unsigned short* __restrict__ Wt,
    int Krows, int N) {
  int stride = gridDim.x * blockDim.x;
  int total = Krows * N;
  for (int i = blockIdx.x * blockDim.x + threadIdx.x; i < total; i += stride) {
    int k = i / N;               // consecutive i -> consecutive n: coalesced reads
    int n = i - k * N;
    Wt[(size_t)n * Krows + k] = f2bf(W[i]);
  }
}

// ---------------- feature build: inp[b, 8640] in bf16 ----------------
// Row layout: u(128) | per k: v(128), poincare, cos, polar, dot, pop

__global__ __launch_bounds__(256) void feature_kernel(
    const float* __restrict__ emb,     // [B, 128]
    const int*   __restrict__ nidx,    // [B, 64]
    const float* __restrict__ freq,    // [V]
    const float* __restrict__ lut,     // [V, 128]
    unsigned short* __restrict__ inp)  // [B, 8640] bf16
{
  const int b    = blockIdx.x;
  const int lane = threadIdx.x & 31;
  const int wave = threadIdx.x >> 5;

  const float4 u4 = *(const float4*)(emb + (size_t)b * Dn + lane * 4);
  const float squ = wave_reduce_add(u4.x*u4.x + u4.y*u4.y + u4.z*u4.z + u4.w*u4.w);

  unsigned short* row = inp + (size_t)b * INn;
  if (wave == 0) {
    row[lane*4+0] = f2bf(u4.x); row[lane*4+1] = f2bf(u4.y);
    row[lane*4+2] = f2bf(u4.z); row[lane*4+3] = f2bf(u4.w);
  }

  for (int k = wave; k < Kn; k += 8) {
    const int idx = nidx[b * Kn + k];
    const float4 v4 = *(const float4*)(lut + (size_t)idx * Dn + lane * 4);

    float dx = u4.x - v4.x, dy = u4.y - v4.y, dz = u4.z - v4.z, dw = u4.w - v4.w;
    float dot = wave_reduce_add(u4.x*v4.x + u4.y*v4.y + u4.z*v4.z + u4.w*v4.w);
    float sqv = wave_reduce_add(v4.x*v4.x + v4.y*v4.y + v4.z*v4.z + v4.w*v4.w);
    float sqd = wave_reduce_add(dx*dx + dy*dy + dz*dz + dw*dw);

    unsigned short* dst = row + Dn + k * (Dn + 5);
    dst[lane*4+0] = f2bf(v4.x); dst[lane*4+1] = f2bf(v4.y);
    dst[lane*4+2] = f2bf(v4.z); dst[lane*4+3] = f2bf(v4.w);

    if (lane == 0) {
      float x = 1.0f + 2.0f * sqd / ((1.0f - squ) * (1.0f - sqv));
      x = fmaxf(x, 1.0f + EPSf);
      float poincare = logf(x + sqrtf(x * x - 1.0f));
      float cosv  = dot / fmaxf(sqrtf(squ) * sqrtf(sqv), 1e-8f);
      float polar = ((squ + sqv + 2.0f * dot) - sqd) * 0.25f;
      float pop   = log1pf(freq[idx]);
      dst[128] = f2bf(poincare);
      dst[129] = f2bf(cosv);
      dst[130] = f2bf(polar);
      dst[131] = f2bf(dot);
      dst[132] = f2bf(pop);
    }
  }
}

// ---------------- WMMA bf16 GEMM: C = act(A @ Bt^T + bias) ----------------
// A:  [M, Ka] bf16 row-major.   Bt: [N, Ka] bf16 row-major (pre-transposed W).
// 256 threads = 8 waves (WR x WC); each wave owns MI x NJ 16x16 output tiles.
// BK=32 -> one v_wmma_f32_16x16x32_bf16 per (i,j) per stage.
// Double-buffered LDS; tiles staged with global_load_async_to_lds_b128.

template<int BM, int BN, int WR, int WC, int MI, int NJ, bool RELU, bool OUT_BF16>
__global__ __launch_bounds__(256) void gemm_bf16_kernel(
    const unsigned short* __restrict__ A,
    const unsigned short* __restrict__ Bt,
    const float* __restrict__ bias,
    void* __restrict__ Cout,
    int M, int N, int Ka)
{
  constexpr int BK  = 32;
  constexpr int CHA = BM / 64;            // 16B async chunks per thread (A tile)
  constexpr int CHB = BN / 64;            // 16B async chunks per thread (B tile)
  constexpr int CHT = CHA + CHB;          // outstanding asyncs per stage per thread

  __shared__ unsigned short lA[2][BM * BK];
  __shared__ unsigned short lB[2][BN * BK];

  const int tid   = threadIdx.x;
  const int lane  = tid & 31;
  const int wv    = tid >> 5;
  const int wr    = wv / WC;
  const int wc    = wv % WC;
  const int bm    = blockIdx.y * BM;
  const int bn    = blockIdx.x * BN;
  const int lr    = lane & 15;
  const int kHalf = lane >> 4;

  // per-thread async chunk descriptors (byte offsets; +kt*64 per stage)
  unsigned aLds[CHA], aOff[CHA];
  #pragma unroll
  for (int ch = 0; ch < CHA; ++ch) {
    int i = tid + ch * 256;               // chunk id in [0, BM*4)
    int r = i >> 2, c = i & 3;            // 4 x 16B chunks per 64B row
    aLds[ch] = (unsigned)(r * 64 + c * 16);
    aOff[ch] = (unsigned)((bm + r) * Ka) * 2u + (unsigned)(c * 16);
  }
  unsigned bLds[CHB], bOff[CHB];
  #pragma unroll
  for (int ch = 0; ch < CHB; ++ch) {
    int i = tid + ch * 256;
    int r = i >> 2, c = i & 3;
    bLds[ch] = (unsigned)(r * 64 + c * 16);
    bOff[ch] = (unsigned)((bn + r) * Ka) * 2u + (unsigned)(c * 16);
  }
  const unsigned ldsA0 = (unsigned)(uintptr_t)&lA[0][0];
  const unsigned ldsA1 = (unsigned)(uintptr_t)&lA[1][0];
  const unsigned ldsB0 = (unsigned)(uintptr_t)&lB[0][0];
  const unsigned ldsB1 = (unsigned)(uintptr_t)&lB[1][0];

  v8f acc[MI][NJ];
  const v8f vzero = {0.f,0.f,0.f,0.f,0.f,0.f,0.f,0.f};
  #pragma unroll
  for (int i = 0; i < MI; ++i)
    #pragma unroll
    for (int j = 0; j < NJ; ++j) acc[i][j] = vzero;

  const int nKT = Ka / BK;

  auto issue_stage = [&](int buf, int kt) {
    const unsigned kb = (unsigned)kt * 64u;
    const unsigned baseA = buf ? ldsA1 : ldsA0;
    const unsigned baseB = buf ? ldsB1 : ldsB0;
    #pragma unroll
    for (int ch = 0; ch < CHA; ++ch)
      async_copy_b128(baseA + aLds[ch], aOff[ch] + kb, (const void*)A);
    #pragma unroll
    for (int ch = 0; ch < CHB; ++ch)
      async_copy_b128(baseB + bLds[ch], bOff[ch] + kb, (const void*)Bt);
  };

  issue_stage(0, 0);

  for (int kt = 0; kt < nKT; ++kt) {
    const int cur = kt & 1;
    if (kt + 1 < nKT) {
      issue_stage(cur ^ 1, kt + 1);
      wait_asynccnt<CHT>();               // stage kt resident (in-order completion)
    } else {
      wait_asynccnt<0>();
    }
    __syncthreads();

    const unsigned short* cA = lA[cur];
    const unsigned short* cB = lB[cur];

    union Frag { v16bf v; unsigned u[8]; };
    Frag afr[MI], bfr[NJ];
    // A fragment (16x32 bf16): lanes 0-15 K{0-7,16-23}, lanes 16-31 K{8-15,24-31}
    #pragma unroll
    for (int i = 0; i < MI; ++i) {
      int m = wr * (MI * 16) + i * 16 + lr;
      #pragma unroll
      for (int j = 0; j < 8; ++j) {
        int k = ((j & 4) << 2) + kHalf * 8 + (j & 3) * 2;
        afr[i].u[j] = *(const unsigned*)&cA[m * BK + k];
      }
    }
    // B fragment (32x16 bf16): lanes 0-15 K{0-7,16-23}, lanes 16-31 K{8-15,24-31}
    #pragma unroll
    for (int jj = 0; jj < NJ; ++jj) {
      int n = wc * (NJ * 16) + jj * 16 + lr;
      #pragma unroll
      for (int j = 0; j < 8; ++j) {
        int k = ((j & 4) << 2) + kHalf * 8 + (j & 3) * 2;
        bfr[jj].u[j] = *(const unsigned*)&cB[n * BK + k];
      }
    }
    #pragma unroll
    for (int i = 0; i < MI; ++i)
      #pragma unroll
      for (int jj = 0; jj < NJ; ++jj)
        acc[i][jj] = __builtin_amdgcn_wmma_f32_16x16x32_bf16(
            false, afr[i].v, false, bfr[jj].v,
            (short)0, acc[i][jj], false, false);
    __syncthreads();
  }

  // epilogue: bias + activation + store
  #pragma unroll
  for (int i = 0; i < MI; ++i) {
    #pragma unroll
    for (int jj = 0; jj < NJ; ++jj) {
      int ng = bn + wc * (NJ * 16) + jj * 16 + lr;
      float bv = bias[ng];
      union { v8f v; float f[8]; } a; a.v = acc[i][jj];
      #pragma unroll
      for (int r = 0; r < 8; ++r) {
        int mg = bm + wr * (MI * 16) + i * 16 + kHalf * 8 + r;
        float val = a.f[r] + bv;
        if (RELU) val = fmaxf(val, 0.f);
        if (OUT_BF16)
          ((unsigned short*)Cout)[(size_t)mg * N + ng] = f2bf(val);
        else
          ((float*)Cout)[(size_t)mg * N + ng] = val;
      }
    }
  }
}

// ---------------- sigmoid head + BCE partial sums ----------------

__global__ __launch_bounds__(256) void head_kernel(
    const float* __restrict__ logits, const float* __restrict__ tgt,
    float* __restrict__ dist, float* __restrict__ partial)
{
  float s = 0.f;
  #pragma unroll
  for (int it = 0; it < 4; ++it) {
    int i = blockIdx.x * 1024 + it * 256 + threadIdx.x;
    float l = logits[i];
    dist[i] = 1.0f / (1.0f + expf(-l));
    float t = tgt[i];
    s += fmaxf(l, 0.f) - l * t + log1pf(expf(-fabsf(l)));
  }
  s = wave_reduce_add(s);
  __shared__ float wsum[8];
  if ((threadIdx.x & 31) == 0) wsum[threadIdx.x >> 5] = s;
  __syncthreads();
  if (threadIdx.x == 0) {
    float tot = 0.f;
    #pragma unroll
    for (int w = 0; w < 8; ++w) tot += wsum[w];
    partial[blockIdx.x] = tot;
  }
}

__global__ __launch_bounds__(256) void loss_reduce_kernel(
    const float* __restrict__ partial, float* __restrict__ out_loss)
{
  float s = partial[threadIdx.x];
  s = wave_reduce_add(s);
  __shared__ float wsum[8];
  if ((threadIdx.x & 31) == 0) wsum[threadIdx.x >> 5] = s;
  __syncthreads();
  if (threadIdx.x == 0) {
    float tot = 0.f;
    #pragma unroll
    for (int w = 0; w < 8; ++w) tot += wsum[w];
    *out_loss = tot / (float)(Bn * Kn);
  }
}

// ---------------- launcher ----------------

extern "C" void kernel_launch(void* const* d_in, const int* in_sizes, int n_in,
                              void* d_out, int out_size, void* d_ws, size_t ws_size,
                              hipStream_t stream) {
  const float* emb  = (const float*)d_in[0];
  const int*   nidx = (const int*)  d_in[1];
  const float* tgt  = (const float*)d_in[2];
  const float* freq = (const float*)d_in[3];
  const float* lut  = (const float*)d_in[4];
  const float* W1   = (const float*)d_in[5];
  const float* b1   = (const float*)d_in[6];
  const float* Wh1  = (const float*)d_in[7];
  const float* bh1  = (const float*)d_in[8];
  const float* Wh2  = (const float*)d_in[9];
  const float* bh2  = (const float*)d_in[10];
  const float* W2   = (const float*)d_in[11];
  const float* b2   = (const float*)d_in[12];

  char* ws = (char*)d_ws;
  size_t off = 0;
  auto carve = [&](size_t bytes) {
    char* p = ws + off;
    off += (bytes + 255) & ~(size_t)255;
    return p;
  };
  unsigned short* inp_bf = (unsigned short*)carve((size_t)Bn * INn * 2);
  unsigned short* W1t    = (unsigned short*)carve((size_t)INn * Hn * 2);  // [H, IN]
  unsigned short* Wh1t   = (unsigned short*)carve((size_t)Hn * Hn * 2);   // [H, H]
  unsigned short* Wh2t   = (unsigned short*)carve((size_t)Hn * Hn * 2);
  unsigned short* W2t    = (unsigned short*)carve((size_t)Hn * Kn * 2);   // [K, H]
  unsigned short* h1bf   = (unsigned short*)carve((size_t)Bn * Hn * 2);
  unsigned short* h2bf   = (unsigned short*)carve((size_t)Bn * Hn * 2);
  unsigned short* h3bf   = (unsigned short*)carve((size_t)Bn * Hn * 2);
  float*          logits = (float*)carve((size_t)Bn * Kn * 4);
  float*          part   = (float*)carve(256 * 4);

  float* dist     = (float*)d_out;
  float* loss_out = (float*)d_out + (size_t)Bn * Kn;

  // weight conversion + transpose: W[k][n] f32 -> Wt[n][k] bf16
  cvt_transpose_kernel<<<2048, 256, 0, stream>>>(W1,  W1t,  INn, Hn);
  cvt_transpose_kernel<<<1024, 256, 0, stream>>>(Wh1, Wh1t, Hn,  Hn);
  cvt_transpose_kernel<<<1024, 256, 0, stream>>>(Wh2, Wh2t, Hn,  Hn);
  cvt_transpose_kernel<<<128,  256, 0, stream>>>(W2,  W2t,  Hn,  Kn);

  // feature build
  feature_kernel<<<Bn, 256, 0, stream>>>(emb, nidx, freq, lut, inp_bf);

  // layer 1: [4096,8640]@[8640,2048] + relu
  gemm_bf16_kernel<128,128,2,4,4,2,true,true>
      <<<dim3(Hn/128, Bn/128), 256, 0, stream>>>(inp_bf, W1t, b1, h1bf, Bn, Hn, INn);
  // layer 2
  gemm_bf16_kernel<128,128,2,4,4,2,true,true>
      <<<dim3(Hn/128, Bn/128), 256, 0, stream>>>(h1bf, Wh1t, bh1, h2bf, Bn, Hn, Hn);
  // layer 3
  gemm_bf16_kernel<128,128,2,4,4,2,true,true>
      <<<dim3(Hn/128, Bn/128), 256, 0, stream>>>(h2bf, Wh2t, bh2, h3bf, Bn, Hn, Hn);
  // layer 4: [4096,2048]@[2048,64], fp32 logits
  gemm_bf16_kernel<128,64,4,2,2,2,false,false>
      <<<dim3(Kn/64, Bn/128), 256, 0, stream>>>(h3bf, W2t, b2, logits, Bn, Kn, Hn);

  // sigmoid + loss
  head_kernel<<<256, 256, 0, stream>>>(logits, tgt, dist, part);
  loss_reduce_kernel<<<1, 256, 0, stream>>>(part, loss_out);
}
